// GNN_SR_Net_49976239456359
// MI455X (gfx1250) — matrix operand order repaired and checked by reading, not compiled, not run on hardware
//
#include <hip/hip_runtime.h>
#include <hip/hip_bf16.h>
#include <math.h>

typedef __attribute__((ext_vector_type(16))) _Float16 v16h;
typedef __attribute__((ext_vector_type(8)))  _Float16 v8h;
typedef __attribute__((ext_vector_type(8)))  float    v8f;

#define NNODE  150000
#define DIM    64
#define NRELS  8
#define NBAS   4
#define TD     384
#define BATCH  1024
#define SEQ    20
#define NPRED  20
#define NEDGE  1000000
#define NSEDGE 200000

// ------------------------------------------------------------------
// small setup / conversion kernels
// ------------------------------------------------------------------
__global__ void k_to_half(const float* __restrict__ in, _Float16* __restrict__ out, int n) {
  int i = blockIdx.x * blockDim.x + threadIdx.x;
  if (i < n) out[i] = (_Float16)in[i];
}

// Wt[r][e][d] = (r<8) ? sum_b comp[r,b]*basis[b,d,e]  : root[d,e]   (f16, transposed)
__global__ void k_make_wt(const float* __restrict__ basis, const float* __restrict__ comp,
                          const float* __restrict__ root, _Float16* __restrict__ Wt) {
  int idx = blockIdx.x * blockDim.x + threadIdx.x;
  if (idx >= 9 * DIM * DIM) return;
  int r = idx >> 12;
  int rem = idx & 4095;
  int e = rem >> 6, d = rem & 63;
  float acc;
  if (r < NRELS) {
    acc = 0.f;
#pragma unroll
    for (int b = 0; b < NBAS; ++b)
      acc += comp[r * NBAS + b] * basis[((size_t)b * DIM + d) * DIM + e];
  } else {
    acc = root[d * DIM + e];
  }
  Wt[((size_t)r * DIM + e) * DIM + d] = (_Float16)acc;
}

// Wt[m][o][i] = W_m[i][o]  (f16, transposed), m in {Q,K,V}
__global__ void k_make_wqkvt(const float* __restrict__ WQ, const float* __restrict__ WK,
                             const float* __restrict__ WV, _Float16* __restrict__ Wt) {
  int idx = blockIdx.x * blockDim.x + threadIdx.x;
  if (idx >= 3 * TD * TD) return;
  int m = idx / (TD * TD);
  int rem = idx - m * (TD * TD);
  int o = rem / TD, i = rem - o * TD;
  const float* W = (m == 0) ? WQ : ((m == 1) ? WK : WV);
  Wt[idx] = (_Float16)W[(size_t)i * TD + o];
}

// ------------------------------------------------------------------
// WMMA GEMM:  C[batch] = A (f16 [M][K], shared) x Bt[batch] (f16 [Ncols][K])
// one 16x16 output tile per wave; K stepped by 32 with v_wmma_f32_16x16x32_f16
// A fragment: lane L holds row M=L&15, K = 16*(L>=16)+j.
//   B fragment: lane L holds col N=L&15, K = 16*(L>=16)+j.
// D fragment: acc[v] = D[v + 8*(L>=16)][L&15].
// ------------------------------------------------------------------
template <int HALF_OUT>
__global__ void k_gemm16(const _Float16* __restrict__ A, const _Float16* __restrict__ Bt,
                         void* __restrict__ Cv, int M, int K, int Ncols, int totalTiles) {
  int wave = blockIdx.x * (blockDim.x >> 5) + (threadIdx.x >> 5);
  if (wave >= totalTiles) return;        // wave-uniform branch; EXEC stays all-1 for WMMA
  int lane = threadIdx.x & 31;
  int ntiles = Ncols >> 4;
  int mtile = wave / ntiles, ntile = wave - mtile * ntiles;
  size_t batch = blockIdx.y;

  const _Float16* Bb = Bt + batch * (size_t)Ncols * K;
  int r16 = lane & 15;
  int hi = lane >> 4;
  const _Float16* arow = A + (size_t)(mtile * 16 + r16) * K + hi * 16;
  const _Float16* brow = Bb + (size_t)(ntile * 16 + r16) * K + hi * 16;

  v8f acc = {};
  for (int k0 = 0; k0 < K; k0 += 32) {
    __builtin_prefetch(arow + k0 + 32, 0, 0);   // global_prefetch_b8
    __builtin_prefetch(brow + k0 + 32, 0, 0);
    v8h alo = *(const v8h*)(arow + k0);
    v8h ahi = *(const v8h*)(arow + k0 + 8);
    v16h b = *(const v16h*)(brow + k0);
    v16h a;
#pragma unroll
    for (int j = 0; j < 8; ++j) { a[j] = alo[j]; a[8 + j] = ahi[j]; }
    acc = __builtin_amdgcn_wmma_f32_16x16x32_f16(false, a, false, b, (short)0, acc,
                                                 false, false);
  }

  size_t cbase = batch * (size_t)M * Ncols + (size_t)ntile * 16 + r16;
  if (HALF_OUT) {
    _Float16* C = (_Float16*)Cv;
#pragma unroll
    for (int v = 0; v < 8; ++v)
      C[cbase + (size_t)(mtile * 16 + v + 8 * hi) * Ncols] = (_Float16)acc[v];
  } else {
    float* C = (float*)Cv;
#pragma unroll
    for (int v = 0; v < 8; ++v)
      C[cbase + (size_t)(mtile * 16 + v + 8 * hi) * Ncols] = acc[v];
  }
}

// ------------------------------------------------------------------
// RGCN aggregation
// ------------------------------------------------------------------
// agg[n][c] = (float)xw_root[n][c] + bias[c]    (fuses root-transform + bias init)
__global__ void k_agg_init(const _Float16* __restrict__ xw_root, const float* __restrict__ bias,
                           float* __restrict__ agg) {
  int idx = blockIdx.x * blockDim.x + threadIdx.x;
  if (idx >= NNODE * DIM) return;
  agg[idx] = (float)xw_root[idx] + bias[idx & 63];
}

// agg[dst] += xw[et][src]   (segment_sum via f32 atomics; HBM/atomic-bound)
__global__ void k_scatter(const _Float16* __restrict__ xw, const int* __restrict__ src,
                          const int* __restrict__ dst, const int* __restrict__ et,
                          float* __restrict__ agg, int E) {
  int idx = blockIdx.x * blockDim.x + threadIdx.x;
  if (idx >= E * 64) return;
  int e = idx >> 6, c = idx & 63;
  int r = et[e], s = src[e], d = dst[e];
  float v = (float)xw[(((size_t)r * NNODE + s) << 6) + c];
  atomicAdd(&agg[((size_t)d << 6) + c], v);
}

// x = tanh(agg); write x (f16, next layer input) and cs[:, l*64 .. ] (f32)
__global__ void k_tanh_cs(const float* __restrict__ agg, _Float16* __restrict__ x_h,
                          float* __restrict__ cs, int layer) {
  int idx = blockIdx.x * blockDim.x + threadIdx.x;
  if (idx >= NNODE * DIM) return;
  int n = idx >> 6, c = idx & 63;
  float v = tanhf(agg[idx]);
  x_h[idx] = (_Float16)v;
  cs[(size_t)n * TD + layer * DIM + c] = v;
}

// ------------------------------------------------------------------
// gathers
// ------------------------------------------------------------------
__global__ void k_gather_user(const float* __restrict__ cs, const int* __restrict__ users,
                              float* __restrict__ out) {
  int idx = blockIdx.x * blockDim.x + threadIdx.x;
  if (idx >= BATCH * TD) return;
  int b = idx / TD, d = idx - b * TD;
  out[idx] = cs[(size_t)users[b] * TD + d];
}

__global__ void k_gather_seq(const float* __restrict__ cs, const int* __restrict__ seqs,
                             float* __restrict__ conv_out, _Float16* __restrict__ seqh) {
  int idx = blockIdx.x * blockDim.x + threadIdx.x;
  if (idx >= BATCH * SEQ * TD) return;
  int s = idx / TD, d = idx - s * TD;
  float v = cs[(size_t)seqs[s] * TD + d];
  conv_out[idx] = v;
  seqh[idx] = (_Float16)v;
}

// ------------------------------------------------------------------
// causal attention with softmax over the QUERY axis (axis=1), per reference
// ------------------------------------------------------------------
__global__ void k_attention(const float* __restrict__ QKV, float* __restrict__ out) {
  __shared__ float sS[SEQ * SEQ];
  int b = blockIdx.x, tid = threadIdx.x;
  const float* Q = QKV + (size_t)b * SEQ * TD;
  const float* Km = QKV + (size_t)BATCH * SEQ * TD + (size_t)b * SEQ * TD;
  const float* V = QKV + 2 * (size_t)BATCH * SEQ * TD + (size_t)b * SEQ * TD;
  const float scale = rsqrtf((float)SEQ);
  for (int idx = tid; idx < SEQ * SEQ; idx += blockDim.x) {
    int i = idx / SEQ, j = idx - i * SEQ;
    float s;
    if (j > i) {
      s = -4294967295.0f;   // NEG = -(2^32)+1
    } else {
      s = 0.f;
      for (int d = 0; d < TD; ++d) s += Q[i * TD + d] * Km[j * TD + d];
      s *= scale;
    }
    sS[idx] = s;
  }
  __syncthreads();
  if (tid < SEQ) {           // softmax over i (query axis) for fixed column j
    int j = tid;
    float m = -INFINITY;
    for (int i = 0; i < SEQ; ++i) m = fmaxf(m, sS[i * SEQ + j]);
    float sum = 0.f;
    for (int i = 0; i < SEQ; ++i) sum += expf(sS[i * SEQ + j] - m);
    float inv = 1.f / sum;
    for (int i = 0; i < SEQ; ++i) sS[i * SEQ + j] = expf(sS[i * SEQ + j] - m) * inv;
  }
  __syncthreads();
  for (int idx = tid; idx < SEQ * TD; idx += blockDim.x) {
    int i = idx / TD, d = idx - i * TD;
    float acc = 0.f;
    for (int j = 0; j < SEQ; ++j) acc += sS[i * SEQ + j] * V[j * TD + d];
    out[((size_t)b * SEQ + i) * TD + d] = acc;
  }
}

// u2[b][d] = user_emb[b][d] + sum_l attn_out[b][l][d]
__global__ void k_u2(const float* __restrict__ user, const float* __restrict__ attn,
                     float* __restrict__ u2) {
  int idx = blockIdx.x * blockDim.x + threadIdx.x;
  if (idx >= BATCH * TD) return;
  int b = idx / TD;
  float s = user[idx];
  const float* ia = attn + (size_t)b * SEQ * TD + (idx - b * TD);
  for (int l = 0; l < SEQ; ++l) s += ia[l * TD];
  u2[idx] = s;
}

// res[b][k] = predict_w[item] . u2[b] + predict_b[item]   (wave-parallel dot, wave32)
__global__ void k_predict(const float* __restrict__ pw, const float* __restrict__ pb,
                          const int* __restrict__ items, const float* __restrict__ u2,
                          float* __restrict__ res) {
  int wave = blockIdx.x * (blockDim.x >> 5) + (threadIdx.x >> 5);
  if (wave >= BATCH * NPRED) return;
  int lane = threadIdx.x & 31;
  int b = wave / NPRED;
  int item = items[wave];
  const float* w = pw + (size_t)item * TD;
  const float* u = u2 + (size_t)b * TD;
  float acc = 0.f;
  for (int c = lane; c < TD; c += 32) acc += w[c] * u[c];
#pragma unroll
  for (int m = 16; m > 0; m >>= 1) acc += __shfl_xor(acc, m, 32);
  if (lane == 0) res[wave] = acc + pb[item];
}

// ------------------------------------------------------------------
extern "C" void kernel_launch(void* const* d_in, const int* in_sizes, int n_in,
                              void* d_out, int out_size, void* d_ws, size_t ws_size,
                              hipStream_t stream) {
  (void)in_sizes; (void)n_in; (void)out_size; (void)ws_size;
  const float* node_emb  = (const float*)d_in[0];
  const float* predict_w = (const float*)d_in[1];
  const float* predict_b = (const float*)d_in[2];
  const float* basis     = (const float*)d_in[3];
  const float* comp      = (const float*)d_in[4];
  const float* root      = (const float*)d_in[5];
  const float* bias      = (const float*)d_in[6];
  const float* sbasis    = (const float*)d_in[7];
  const float* scomp     = (const float*)d_in[8];
  const float* sroot     = (const float*)d_in[9];
  const float* sbias     = (const float*)d_in[10];
  const float* WQ        = (const float*)d_in[11];
  const float* WK        = (const float*)d_in[12];
  const float* WV        = (const float*)d_in[13];
  const int* users = (const int*)d_in[14];
  const int* seqs  = (const int*)d_in[15];
  const int* items = (const int*)d_in[16];
  const int* ei    = (const int*)d_in[17];
  const int* et    = (const int*)d_in[18];
  const int* sei   = (const int*)d_in[20];
  const int* setp  = (const int*)d_in[21];

  // workspace carving (all regions fully written before read; ~605 MB)
  char* ws = (char*)d_ws;
  size_t off = 0;
  auto carve = [&](size_t bytes) -> char* {
    char* p = ws + off;
    off = (off + bytes + 255) & ~(size_t)255;
    return p;
  };
  _Float16* x_h   = (_Float16*)carve((size_t)NNODE * DIM * 2);
  _Float16* Wt    = (_Float16*)carve((size_t)9 * DIM * DIM * 2);
  _Float16* xw    = (_Float16*)carve((size_t)9 * NNODE * DIM * 2);
  float*    agg   = (float*)carve((size_t)NNODE * DIM * 4);
  float*    cs    = (float*)carve((size_t)NNODE * TD * 4);
  _Float16* wqkvt = (_Float16*)carve((size_t)3 * TD * TD * 2);
  _Float16* seqh  = (_Float16*)carve((size_t)BATCH * SEQ * TD * 2);
  float*    qkv   = (float*)carve((size_t)3 * BATCH * SEQ * TD * 4);
  float*    attn  = (float*)carve((size_t)BATCH * SEQ * TD * 4);
  float*    u2    = (float*)carve((size_t)BATCH * TD * 4);

  float* res_out  = (float*)d_out;                 // [1024, 20]
  float* user_out = res_out + BATCH * NPRED;       // [1024, 384]
  float* conv_out = user_out + BATCH * TD;         // [1024, 20, 384]

  k_to_half<<<(NNODE * DIM + 255) / 256, 256, 0, stream>>>(node_emb, x_h, NNODE * DIM);

  for (int l = 0; l < 6; ++l) {
    bool fullg = l < 3;
    int ll = fullg ? l : l - 3;
    const float* bas = (fullg ? basis : sbasis) + (size_t)ll * NBAS * DIM * DIM;
    const float* cm  = (fullg ? comp : scomp) + (size_t)ll * NRELS * NBAS;
    const float* rt  = (fullg ? root : sroot) + (size_t)ll * DIM * DIM;
    const float* bi  = (fullg ? bias : sbias) + (size_t)ll * DIM;
    const int* esrc  = fullg ? ei : sei;
    const int* edst  = fullg ? (ei + NEDGE) : (sei + NSEDGE);
    const int* etyp  = fullg ? et : setp;
    int E = fullg ? NEDGE : NSEDGE;

    k_make_wt<<<(9 * DIM * DIM + 255) / 256, 256, 0, stream>>>(bas, cm, rt, Wt);
    int tiles = (NNODE / 16) * (DIM / 16);  // 37500 16x16 tiles per relation
    k_gemm16<1><<<dim3((tiles + 3) / 4, 9), 128, 0, stream>>>(x_h, Wt, xw, NNODE, DIM, DIM,
                                                              tiles);
    k_agg_init<<<(NNODE * DIM + 255) / 256, 256, 0, stream>>>(xw + (size_t)NRELS * NNODE * DIM,
                                                              bi, agg);
    k_scatter<<<(E * 64 + 255) / 256, 256, 0, stream>>>(xw, esrc, edst, etyp, agg, E);
    k_tanh_cs<<<(NNODE * DIM + 255) / 256, 256, 0, stream>>>(agg, x_h, cs, l);
  }

  k_make_wqkvt<<<(3 * TD * TD + 255) / 256, 256, 0, stream>>>(WQ, WK, WV, wqkvt);
  k_gather_user<<<(BATCH * TD + 255) / 256, 256, 0, stream>>>(cs, users, user_out);
  k_gather_seq<<<(BATCH * SEQ * TD + 255) / 256, 256, 0, stream>>>(cs, seqs, conv_out, seqh);

  int qtiles = (BATCH * SEQ / 16) * (TD / 16);  // 30720 tiles per {Q,K,V}
  k_gemm16<0><<<dim3((qtiles + 3) / 4, 3), 128, 0, stream>>>(seqh, wqkvt, qkv, BATCH * SEQ, TD,
                                                             TD, qtiles);
  k_attention<<<BATCH, 256, 0, stream>>>(qkv, attn);
  k_u2<<<(BATCH * TD + 255) / 256, 256, 0, stream>>>(user_out, attn, u2);
  k_predict<<<(BATCH * NPRED + 7) / 8, 256, 0, stream>>>(predict_w, predict_b, items, u2,
                                                         res_out);
}
